// AttentionPairBias_16793322127892
// MI455X (gfx1250) — matrix operand order repaired and checked by reading, not compiled, not run on hardware
//
#include <hip/hip_runtime.h>
#include <hip/hip_bf16.h>

// ---------- WMMA bf16 helpers (CDNA5 gfx1250, wave32) ----------
typedef __attribute__((ext_vector_type(16))) __bf16 v16bf;
typedef __attribute__((ext_vector_type(8)))  float  v8f;

#define DEV static __device__ __forceinline__

union Frag16 { v16bf v; uint4 q[2]; };
union Pack8  { __bf16 h[8]; uint4 q; };

// A-fragment (16x32 bf16, M x K): lane M = lane%16, half = lane/16.
// Per lane: K = half*8 + [0..7]  then  K = 16 + half*8 + [0..7]   (two 16B chunks)
DEV v16bf frag_a(const __bf16* rowbase, int k0, int half) {
  Frag16 u;
  u.q[0] = *reinterpret_cast<const uint4*>(rowbase + k0 + half * 8);
  u.q[1] = *reinterpret_cast<const uint4*>(rowbase + k0 + 16 + half * 8);
  return u.v;
}
// B-fragment (32x16 bf16, K x N) from B^T stored [N][K]: lane N = lane%16,
// per lane: K = half*8 + [0..7]  then  K = 16 + half*8 + [0..7]   (two 16B chunks)
DEV v16bf frag_b(const __bf16* colbase, int k0, int half) {
  Frag16 u;
  u.q[0] = *reinterpret_cast<const uint4*>(colbase + k0 + half * 8);
  u.q[1] = *reinterpret_cast<const uint4*>(colbase + k0 + 16 + half * 8);
  return u.v;
}
DEV v8f wmma_bf16(v16bf a, v16bf b, v8f c) {
  return __builtin_amdgcn_wmma_f32_16x16x32_bf16(false, a, false, b, (short)0, c,
                                                 false, false);
}

// ---------- constants ----------
// B=1, N=768, D=768, H=16, HD=48 (padded to 64), ZD=128
#define NN   768
#define DD   768
#define HH   16
#define HDP  64           // padded head dim
#define DP   1024         // padded Q/K row length = H*HDP
#define NT48 48           // 768/16 tiles

// ============================================================
// Kernel 1: LayerNorm(s) -> sn (bf16, row major [768][768])
// ============================================================
__global__ void __launch_bounds__(256)
k_ln_s(const float* __restrict__ s, const float* __restrict__ w,
       const float* __restrict__ b, __bf16* __restrict__ sn) {
  const int row = blockIdx.x, t = threadIdx.x;
  const float* sr = s + (size_t)row * DD;
  float x0 = sr[t], x1 = sr[t + 256], x2 = sr[t + 512];
  float sum = x0 + x1 + x2;
  float ss  = x0 * x0 + x1 * x1 + x2 * x2;
#pragma unroll
  for (int o = 16; o > 0; o >>= 1) {
    sum += __shfl_xor(sum, o, 32);
    ss  += __shfl_xor(ss, o, 32);
  }
  __shared__ float rs[8], rq[8];
  const int wv = t >> 5, lane = t & 31;
  if (lane == 0) { rs[wv] = sum; rq[wv] = ss; }
  __syncthreads();
  if (t == 0) {
    float a = 0.f, c = 0.f;
    for (int i = 0; i < 8; i++) { a += rs[i]; c += rq[i]; }
    rs[0] = a; rq[0] = c;
  }
  __syncthreads();
  const float mean = rs[0] * (1.0f / 768.0f);
  const float var  = rq[0] * (1.0f / 768.0f) - mean * mean;
  const float r    = rsqrtf(var + 1e-5f);
  __bf16* dr = sn + (size_t)row * DD;
  dr[t]       = (__bf16)((x0 - mean) * r * w[t]       + b[t]);
  dr[t + 256] = (__bf16)((x1 - mean) * r * w[t + 256] + b[t + 256]);
  dr[t + 512] = (__bf16)((x2 - mean) * r * w[t + 512] + b[t + 512]);
}

// ============================================================
// Kernel 2: transpose-convert five [768][768] f32 weights to bf16 [out][in]
// ============================================================
__global__ void __launch_bounds__(1024)
k_wt(const float* __restrict__ q_w, const float* __restrict__ k_w,
     const float* __restrict__ v_w, const float* __restrict__ g_w,
     const float* __restrict__ o_w,
     __bf16* __restrict__ qwt, __bf16* __restrict__ kwt,
     __bf16* __restrict__ vwt, __bf16* __restrict__ gwt,
     __bf16* __restrict__ owt) {
  const float* srcs[5] = {q_w, k_w, v_w, g_w, o_w};
  __bf16* dsts[5] = {qwt, kwt, vwt, gwt, owt};
  const float* src = srcs[blockIdx.z];
  __bf16* dst = dsts[blockIdx.z];
  __shared__ float tile[32][33];
  const int i = blockIdx.y * 32 + threadIdx.y;   // input row (d_in)
  const int o = blockIdx.x * 32 + threadIdx.x;   // output col (d_out)
  tile[threadIdx.y][threadIdx.x] = src[(size_t)i * DD + o];
  __syncthreads();
  const int oo = blockIdx.x * 32 + threadIdx.y;
  const int ii = blockIdx.y * 32 + threadIdx.x;
  dst[(size_t)oo * DD + ii] = (__bf16)tile[threadIdx.x][threadIdx.y];
}

// z_w [128][16] f32 -> zwt [16][128] bf16
__global__ void k_zwt(const float* __restrict__ z_w, __bf16* __restrict__ zwt) {
  const int t = threadIdx.x;
#pragma unroll
  for (int j = 0; j < 8; j++) {
    int idx = t + j * 256;            // 0..2047
    int h = idx & 15, c = idx >> 4;   // c: 0..127
    zwt[h * 128 + c] = (__bf16)z_w[c * HH + h];
  }
}

// ============================================================
// Kernel 3: fused Q/K/V/G GEMM. One wave per 16x16 tile, 4 outputs share A.
//   Qp [768][1024] bf16 (padded, scaled, +q_b), Kp same layout,
//   Vt [768 cols][768 keys] bf16 (transposed), Gb [768][768] bf16 (sigmoid)
// ============================================================
__global__ void __launch_bounds__(128)
k_qkvg(const __bf16* __restrict__ sn,
       const __bf16* __restrict__ qwt, const __bf16* __restrict__ kwt,
       const __bf16* __restrict__ vwt, const __bf16* __restrict__ gwt,
       const float* __restrict__ q_b,
       __bf16* __restrict__ Qp, __bf16* __restrict__ Kp,
       __bf16* __restrict__ Vt, __bf16* __restrict__ Gb) {
  const int wv = threadIdx.x >> 5, lane = threadIdx.x & 31;
  const int wgid = blockIdx.x * 4 + wv;            // 0..2303
  const int mt = wgid / NT48, nt = wgid % NT48;
  const int half = lane >> 4, n = lane & 15;
  const __bf16* arow = sn + (size_t)(mt * 16 + (lane & 15)) * DD;
  const __bf16* bq = qwt + (size_t)(nt * 16 + n) * DD;
  const __bf16* bk = kwt + (size_t)(nt * 16 + n) * DD;
  const __bf16* bv = vwt + (size_t)(nt * 16 + n) * DD;
  const __bf16* bg = gwt + (size_t)(nt * 16 + n) * DD;
  v8f aq = {}, ak = {}, av = {}, ag = {};
  for (int k0 = 0; k0 < DD; k0 += 32) {
    v16bf a = frag_a(arow, k0, half);
    aq = wmma_bf16(a, frag_b(bq, k0, half), aq);
    ak = wmma_bf16(a, frag_b(bk, k0, half), ak);
    av = wmma_bf16(a, frag_b(bv, k0, half), av);
    ag = wmma_bf16(a, frag_b(bg, k0, half), ag);
  }
  const int col = nt * 16 + n;
  const int h = col / 48, dIn = col % 48;
  const float scale = 0.14433756729740643f;   // 1/sqrt(48)
  const float qb = q_b[col];
  __shared__ float vt_tile[4][16][17];
#pragma unroll
  for (int r = 0; r < 8; r++) {
    int M = half * 8 + r;
    int row = mt * 16 + M;
    Qp[(size_t)row * DP + h * HDP + dIn] = (__bf16)((aq[r] + qb) * scale);
    Kp[(size_t)row * DP + h * HDP + dIn] = (__bf16)ak[r];
    Gb[(size_t)row * DD + col] = (__bf16)(1.0f / (1.0f + __expf(-ag[r])));
    vt_tile[wv][n][M] = av[r];
  }
  __syncthreads();
  // V transpose: per-wave LDS tile -> Vt[col][key] coalesced 16B stores
  const int c2 = lane >> 1, kh = lane & 1;
  Pack8 pk;
#pragma unroll
  for (int j = 0; j < 8; j++) pk.h[j] = (__bf16)vt_tile[wv][c2][kh * 8 + j];
  *reinterpret_cast<uint4*>(Vt + (size_t)(nt * 16 + c2) * DD + mt * 16 + kh * 8) = pk.q;
}

// ============================================================
// Kernel 4: pair bias  zb[h][q][k] = LN(z[q,k,:]) @ z_w  (bf16 output)
// Block = 256 thr (8 waves); block handles (q, 128 consecutive k).
// Wave: 16 rows; 2 lanes/row each own 64 channels (registers), LN in-regs,
// normalized bf16 tile -> LDS, 4 WMMA k-steps vs zwt, LDS transpose, store.
// ============================================================
__global__ void __launch_bounds__(256)
k_zbias(const float* __restrict__ z, const float* __restrict__ znw,
        const float* __restrict__ znb, const __bf16* __restrict__ zwt,
        __bf16* __restrict__ zb) {
  const int q = blockIdx.y;
  const int kb = blockIdx.x * 128;
  const int t = threadIdx.x, wv = t >> 5, lane = t & 31;
  __shared__ float lw[128], lb[128];
  __shared__ __attribute__((aligned(16))) __bf16 atile[8][16 * 128];
  __shared__ float ttile[8][16][17];
  if (t < 128) { lw[t] = znw[t]; lb[t] = znb[t]; }
  __syncthreads();

  const int r = lane >> 1, ch0 = (lane & 1) * 64;
  const int krow = kb + wv * 16 + r;
  const float4* zp =
      reinterpret_cast<const float4*>(z + ((size_t)q * NN + krow) * 128 + ch0);
  float4 xv[16];
  float sum = 0.f, ss = 0.f;
#pragma unroll
  for (int j = 0; j < 16; j++) {
    xv[j] = zp[j];
    sum += xv[j].x + xv[j].y + xv[j].z + xv[j].w;
    ss  += xv[j].x * xv[j].x + xv[j].y * xv[j].y + xv[j].z * xv[j].z +
           xv[j].w * xv[j].w;
  }
  sum += __shfl_xor(sum, 1, 32);
  ss  += __shfl_xor(ss, 1, 32);
  const float mean = sum * (1.0f / 128.0f);
  const float var  = ss * (1.0f / 128.0f) - mean * mean;
  const float rstd = rsqrtf(var + 1e-5f);
  __bf16* al = &atile[wv][r * 128 + ch0];
#pragma unroll
  for (int j = 0; j < 8; j++) {
    Pack8 pk;
    const float* xf = reinterpret_cast<const float*>(&xv[j * 2]);
#pragma unroll
    for (int e = 0; e < 8; e++) {
      int c = ch0 + j * 8 + e;
      pk.h[e] = (__bf16)((xf[e] - mean) * rstd * lw[c] + lb[c]);
    }
    reinterpret_cast<uint4*>(al)[j] = pk.q;
  }
  __syncthreads();

  const int half = lane >> 4, n = lane & 15;
  const __bf16* ar = &atile[wv][(lane & 15) * 128];
  const __bf16* br = zwt + n * 128;
  v8f acc = {};
#pragma unroll
  for (int ks = 0; ks < 4; ks++)
    acc = wmma_bf16(frag_a(ar, ks * 32, half), frag_b(br, ks * 32, half), acc);
#pragma unroll
  for (int rr = 0; rr < 8; rr++) ttile[wv][n][half * 8 + rr] = acc[rr];
  __syncthreads();

  const int h2 = lane >> 1, kh = lane & 1;
  Pack8 pk;
#pragma unroll
  for (int j = 0; j < 8; j++) pk.h[j] = (__bf16)ttile[wv][h2][kh * 8 + j];
  *reinterpret_cast<uint4*>(zb + (size_t)h2 * (NN * NN) + (size_t)q * NN + kb +
                            wv * 16 + kh * 8) = pk.q;
}

// ============================================================
// Kernel 5: attention. One block (4 waves) per (q-tile, head).
// scores (WMMA, K=64 padded) + bias -> LDS f32, softmax, probs bf16 -> LDS,
// A@V (WMMA over 768 keys), multiply by gate, write og bf16 [q][col].
// ============================================================
__global__ void __launch_bounds__(128)
k_attn(const __bf16* __restrict__ Qp, const __bf16* __restrict__ Kp,
       const __bf16* __restrict__ Vt, const __bf16* __restrict__ Gb,
       const __bf16* __restrict__ zb, __bf16* __restrict__ og) {
  const int qt = blockIdx.x, h = blockIdx.y;
  const int wv = threadIdx.x >> 5, lane = threadIdx.x & 31;
  const int half = lane >> 4, n = lane & 15;
  __shared__ float sc[16][768];                               // 48 KB
  __shared__ __attribute__((aligned(16))) __bf16 pb[16][768]; // 24 KB

  // phase 1: scores for 12 key-tiles per wave
  const __bf16* qbase = Qp + (size_t)(qt * 16 + (lane & 15)) * DP + h * HDP;
  v16bf a0 = frag_a(qbase, 0, half);
  v16bf a1 = frag_a(qbase, 32, half);
  const __bf16* zbh = zb + (size_t)h * (NN * NN) + (size_t)(qt * 16) * NN;
  for (int kt = wv; kt < NT48; kt += 4) {
    const __bf16* kbase = Kp + (size_t)(kt * 16 + n) * DP + h * HDP;
    v8f acc = {};
    acc = wmma_bf16(a0, frag_b(kbase, 0, half), acc);
    acc = wmma_bf16(a1, frag_b(kbase, 32, half), acc);
#pragma unroll
    for (int r = 0; r < 8; r++) {
      int M = half * 8 + r;
      float bias = (float)zbh[(size_t)M * NN + kt * 16 + n];
      sc[M][kt * 16 + n] = acc[r] + bias;
    }
  }
  __syncthreads();

  // phase 2: softmax, 4 rows per wave
#pragma unroll
  for (int i = 0; i < 4; i++) {
    int row = wv * 4 + i;
    float m = -3.0e38f;
#pragma unroll
    for (int j = 0; j < 24; j++) m = fmaxf(m, sc[row][lane + j * 32]);
#pragma unroll
    for (int o = 16; o > 0; o >>= 1) m = fmaxf(m, __shfl_xor(m, o, 32));
    float sum = 0.f;
#pragma unroll
    for (int j = 0; j < 24; j++) {
      float e = __expf(sc[row][lane + j * 32] - m);
      sc[row][lane + j * 32] = e;
      sum += e;
    }
#pragma unroll
    for (int o = 16; o > 0; o >>= 1) sum += __shfl_xor(sum, o, 32);
    float inv = 1.0f / sum;
#pragma unroll
    for (int j = 0; j < 24; j++)
      pb[row][lane + j * 32] = (__bf16)(sc[row][lane + j * 32] * inv);
  }
  __syncthreads();

  // phase 3: A @ V, 3 column tiles (48 = 3*16), waves 0..2
  if (wv < 3) {
    const __bf16* ap = &pb[lane & 15][0];
    const int col = h * 48 + wv * 16 + n;
    const __bf16* vb = Vt + (size_t)col * DD;
    v8f acc = {};
    for (int k0 = 0; k0 < DD; k0 += 32)
      acc = wmma_bf16(frag_a(ap, k0, half), frag_b(vb, k0, half), acc);
#pragma unroll
    for (int r = 0; r < 8; r++) {
      int row = qt * 16 + half * 8 + r;
      float g = (float)Gb[(size_t)row * DD + col];
      og[(size_t)row * DD + col] = (__bf16)(acc[r] * g);
    }
  }
}

// ============================================================
// Kernel 6: out = og @ o_w  (f32 output)
// ============================================================
__global__ void __launch_bounds__(128)
k_out(const __bf16* __restrict__ og, const __bf16* __restrict__ owt,
      float* __restrict__ out) {
  const int wv = threadIdx.x >> 5, lane = threadIdx.x & 31;
  const int wgid = blockIdx.x * 4 + wv;
  const int mt = wgid / NT48, nt = wgid % NT48;
  const int half = lane >> 4, n = lane & 15;
  const __bf16* arow = og + (size_t)(mt * 16 + (lane & 15)) * DD;
  const __bf16* bcol = owt + (size_t)(nt * 16 + n) * DD;
  v8f acc = {};
  for (int k0 = 0; k0 < DD; k0 += 32)
    acc = wmma_bf16(frag_a(arow, k0, half), frag_b(bcol, k0, half), acc);
#pragma unroll
  for (int r = 0; r < 8; r++)
    out[(size_t)(mt * 16 + half * 8 + r) * DD + nt * 16 + n] = acc[r];
}

// ============================================================
extern "C" void kernel_launch(void* const* d_in, const int* in_sizes, int n_in,
                              void* d_out, int out_size, void* d_ws,
                              size_t ws_size, hipStream_t stream) {
  const float* s   = (const float*)d_in[0];
  const float* z   = (const float*)d_in[1];
  const float* nsw = (const float*)d_in[2];
  const float* nsb = (const float*)d_in[3];
  const float* q_w = (const float*)d_in[4];
  const float* q_b = (const float*)d_in[5];
  const float* k_w = (const float*)d_in[6];
  const float* v_w = (const float*)d_in[7];
  const float* g_w = (const float*)d_in[8];
  const float* znw = (const float*)d_in[9];
  const float* znb = (const float*)d_in[10];
  const float* z_w = (const float*)d_in[11];
  const float* o_w = (const float*)d_in[12];
  float* out = (float*)d_out;

  char* p = (char*)d_ws;
  auto take = [&](size_t bytes) {
    char* r = p;
    p += (bytes + 255) & ~(size_t)255;
    return r;
  };
  __bf16* sn   = (__bf16*)take((size_t)NN * DD * 2);
  __bf16* qwt  = (__bf16*)take((size_t)DD * DD * 2);
  __bf16* kwt  = (__bf16*)take((size_t)DD * DD * 2);
  __bf16* vwt  = (__bf16*)take((size_t)DD * DD * 2);
  __bf16* gwt  = (__bf16*)take((size_t)DD * DD * 2);
  __bf16* owt  = (__bf16*)take((size_t)DD * DD * 2);
  __bf16* zwt  = (__bf16*)take((size_t)HH * 128 * 2);
  __bf16* Qp   = (__bf16*)take((size_t)NN * DP * 2);
  __bf16* Kp   = (__bf16*)take((size_t)NN * DP * 2);
  __bf16* Vt   = (__bf16*)take((size_t)DD * NN * 2);
  __bf16* Gb   = (__bf16*)take((size_t)NN * DD * 2);
  __bf16* ogb  = (__bf16*)take((size_t)NN * DD * 2);
  __bf16* zbuf = (__bf16*)take((size_t)HH * NN * NN * 2);
  (void)ws_size; (void)in_sizes; (void)n_in; (void)out_size;

  // zero the padded columns of Q/K (d 48..63 within each head)
  (void)hipMemsetAsync(Qp, 0, (size_t)NN * DP * 2, stream);
  (void)hipMemsetAsync(Kp, 0, (size_t)NN * DP * 2, stream);

  k_ln_s<<<NN, 256, 0, stream>>>(s, nsw, nsb, sn);
  k_wt<<<dim3(24, 24, 5), dim3(32, 32), 0, stream>>>(
      q_w, k_w, v_w, g_w, o_w, qwt, kwt, vwt, gwt, owt);
  k_zwt<<<1, 256, 0, stream>>>(z_w, zwt);
  k_qkvg<<<576, 128, 0, stream>>>(sn, qwt, kwt, vwt, gwt, q_b, Qp, Kp, Vt, Gb);
  k_zbias<<<dim3(6, NN), 256, 0, stream>>>(z, znw, znb, zwt, zbuf);
  k_attn<<<dim3(NT48, HH), 128, 0, stream>>>(Qp, Kp, Vt, Gb, zbuf, ogb);
  k_out<<<576, 128, 0, stream>>>(ogb, owt, out);
}
